// CrossModalAttention_10307921510532
// MI455X (gfx1250) — matrix operand order repaired and checked by reading, not compiled, not run on hardware
//
#include <hip/hip_runtime.h>
#include <hip/hip_bf16.h>

#define BB 4
#define LL 1024
#define DD 1024
#define HH 16
#define HD 64

typedef __attribute__((ext_vector_type(16))) _Float16 v16h;
typedef __attribute__((ext_vector_type(8)))  _Float16 v8h;
typedef __attribute__((ext_vector_type(4)))  _Float16 v4h;
typedef __attribute__((ext_vector_type(8)))  float    v8f;
typedef __attribute__((ext_vector_type(4)))  unsigned int u32x4;
typedef __attribute__((ext_vector_type(8)))  int      i32x8;
typedef __attribute__((ext_vector_type(4)))  int      i32x4;

#if defined(__has_builtin)
#if __has_builtin(__builtin_amdgcn_tensor_load_to_lds)
#define HAVE_TDM 1
#else
#define HAVE_TDM 0
#endif
#else
#define HAVE_TDM 0
#endif

__device__ inline v8f v8f_zero() {
  v8f z = {0.f, 0.f, 0.f, 0.f, 0.f, 0.f, 0.f, 0.f};
  return z;
}

__device__ inline v16h make_frag(v8h lo, v8h hi) {
  v16h r;
#pragma unroll
  for (int i = 0; i < 8; ++i) { r[i] = lo[i]; r[8 + i] = hi[i]; }
  return r;
}

// A-matrix 16x32 f16 fragment (ISA 7.12.2): lane m = t%16, half hi = t/16.
// elements 0..7 -> K = 8*hi + 0..7 ; elements 8..15 -> K = 16 + 8*hi + 0..7
__device__ inline v16h load_a_frag(const _Float16* base, int ld, int lane) {
  int m = lane & 15, hi = lane >> 4;
  const _Float16* p = base + (size_t)m * ld + 8 * hi;
  v8h lo = *(const v8h*)p;
  v8h hh = *(const v8h*)(p + 16);
  return make_frag(lo, hh);
}

// B-matrix 32x16 f16 fragment from transposed (Bt[n][k] row-major) buffer:
// lane n = t%16, half h2 = t/16; elements 0..7 -> K = 8*h2 + 0..7 ; 8..15 -> K = 16 + 8*h2 + 0..7.
__device__ inline v16h load_bt_frag(const _Float16* base, int ld, int lane) {
  int n = lane & 15, h2 = lane >> 4;
  const _Float16* p = base + (size_t)n * ld + 8 * h2;
  v8h lo = *(const v8h*)p;
  v8h hh = *(const v8h*)(p + 16);
  return make_frag(lo, hh);
}

__device__ inline v8f wmma32(v16h a, v16h b, v8f c) {
  return __builtin_amdgcn_wmma_f32_16x16x32_f16(false, a, false, b, (short)0, c,
                                                false, false);
}

#if HAVE_TDM
// TDM 2D tile load: f16 elements, tile (tile_d0 x tile_d1), row stride
// `stride_elems` in the source tensor, LDS rows padded +16B per 128B row
// (pad_interval code 4 = 32 DWORDs, pad_amount code 3 = 4 DWORDs) so the
// LDS layout is rows of 72 halves (144 B) — conflict-free for b128 frags.
__device__ inline void tdm_load_2d(unsigned lds_addr, const _Float16* gptr,
                                   unsigned tile_d0, unsigned tile_d1,
                                   unsigned long long stride_elems) {
  unsigned long long ga = (unsigned long long)(size_t)(const void*)gptr;
  u32x4 g0;
  g0[0] = 1u;                                             // count=1, user D#
  g0[1] = lds_addr;                                       // lds_addr [63:32]
  g0[2] = (unsigned)(ga & 0xffffffffull);                 // global_addr lo
  g0[3] = (unsigned)((ga >> 32) & 0x01ffffffull) | (2u << 30);  // hi + type=2
  i32x8 g1;
  // data_size=1 (2B), pad_enable, pad_interval=4 (32DW), pad_amount=3 (4DW)
  g1[0] = (int)((1u << 16) | (1u << 20) | (4u << 22) | (3u << 25));
  // tensor_dim0 = 0x40000000 (huge, no OOB clip): bits[79:48]
  g1[1] = 0;                                   // dim0[15:0] << 16 = 0
  g1[2] = (int)0x4000u;                        // dim0[31:16] | dim1[15:0]<<16
  g1[3] = (int)(0x4000u | (tile_d0 << 16));    // dim1[31:16] | tile_dim0
  g1[4] = (int)(tile_d1 & 0xffffu);            // tile_dim1 | tile_dim2(=0)
  g1[5] = (int)(unsigned)(stride_elems & 0xffffffffull);  // dim0_stride lo
  g1[6] = (int)(unsigned)((stride_elems >> 32) & 0xffffull);
  g1[7] = 0;                                   // dim1_stride (unused, 2D)
  i32x4 z4 = {0, 0, 0, 0};
  i32x8 z8 = {0, 0, 0, 0, 0, 0, 0, 0};
  __builtin_amdgcn_tensor_load_to_lds(g0, g1, z4, z4, z8, 0);
}
#endif

// ---------------------------------------------------------------------------
// Elementwise f32 -> f16 convert (activations and weights), 4 elems/thread.
// ---------------------------------------------------------------------------
__global__ void __launch_bounds__(256)
cvt_f16(const float* __restrict__ src, _Float16* __restrict__ dst) {
  int i = blockIdx.x * 256 + threadIdx.x;
  float4 v = ((const float4*)src)[i];
  v4h o;
  o[0] = (_Float16)v.x; o[1] = (_Float16)v.y;
  o[2] = (_Float16)v.z; o[3] = (_Float16)v.w;
  ((v4h*)dst)[i] = o;
}

// ---------------------------------------------------------------------------
// f16 GEMM core: Out[m][n] = sum_k A[m][k]*W[n][k]  (W rows are B^T rows)
// Block 256 (8 waves), tile 128x64, K staged 64-wide, TDM double-buffered.
// MODE 0: Out_h = f16(acc + bias)          (projection)
// MODE 1: Out_f = acc + bias + resid       (output projection + residual)
// ---------------------------------------------------------------------------
template <int MODE>
__device__ inline void gemm_body(const _Float16* __restrict__ A,
                                 const _Float16* __restrict__ W,
                                 const float* __restrict__ bias,
                                 const float* __restrict__ resid,
                                 _Float16* __restrict__ OutH,
                                 float* __restrict__ OutF) {
  __shared__ __align__(16) _Float16 Alds[2][128][72];
  __shared__ __align__(16) _Float16 Blds[2][64][72];
  const int tid = threadIdx.x, lane = tid & 31, w = tid >> 5;
  const int row0 = blockIdx.y * 128, col0 = blockIdx.x * 64;
  const int NKS = DD / 64;

  v8f acc[4];
#pragma unroll
  for (int i = 0; i < 4; ++i) acc[i] = v8f_zero();

#if HAVE_TDM
  if (tid < 32) {
    tdm_load_2d((unsigned)(size_t)(void*)&Alds[0][0][0], A + (size_t)row0 * DD,
                64, 128, DD);
    tdm_load_2d((unsigned)(size_t)(void*)&Blds[0][0][0], W + (size_t)col0 * DD,
                64, 64, DD);
  }
#endif
  for (int s = 0; s < NKS; ++s) {
    int cur = s & 1;
#if HAVE_TDM
    if (tid < 32) __builtin_amdgcn_s_wait_tensorcnt(0);
    __syncthreads();
    if (s + 1 < NKS && tid < 32) {
      int k0n = (s + 1) * 64;
      tdm_load_2d((unsigned)(size_t)(void*)&Alds[cur ^ 1][0][0],
                  A + (size_t)row0 * DD + k0n, 64, 128, DD);
      tdm_load_2d((unsigned)(size_t)(void*)&Blds[cur ^ 1][0][0],
                  W + (size_t)col0 * DD + k0n, 64, 64, DD);
    }
#else
    cur = 0;
    int k0 = s * 64;
    for (int i = tid; i < 128 * 64; i += 256) {
      int r = i >> 6, k = i & 63;
      Alds[0][r][k] = A[(size_t)(row0 + r) * DD + k0 + k];
    }
    for (int i = tid; i < 64 * 64; i += 256) {
      int nn = i >> 6, k = i & 63;
      Blds[0][nn][k] = W[(size_t)(col0 + nn) * DD + k0 + k];
    }
    __syncthreads();
#endif
#pragma unroll
    for (int ks = 0; ks < 2; ++ks) {
      v16h afrag = load_a_frag(&Alds[cur][w * 16][ks * 32], 72, lane);
#pragma unroll
      for (int nt = 0; nt < 4; ++nt) {
        v16h bfrag = load_bt_frag(&Blds[cur][nt * 16][ks * 32], 72, lane);
        acc[nt] = wmma32(afrag, bfrag, acc[nt]);
      }
    }
    __syncthreads();
  }

  const int n = lane & 15, hi = lane >> 4;
#pragma unroll
  for (int nt = 0; nt < 4; ++nt) {
    int col = col0 + nt * 16 + n;
    float bv = bias[col];
#pragma unroll
    for (int r = 0; r < 8; ++r) {
      int row = row0 + w * 16 + r + 8 * hi;
      if (MODE == 0) {
        OutH[(size_t)row * DD + col] = (_Float16)(acc[nt][r] + bv);
      } else {
        OutF[(size_t)row * DD + col] =
            acc[nt][r] + bv + resid[(size_t)row * DD + col];
      }
    }
  }
}

__global__ void __launch_bounds__(256)
proj_gemm(const _Float16* __restrict__ A, const _Float16* __restrict__ W,
          const float* __restrict__ bias, _Float16* __restrict__ Out) {
  gemm_body<0>(A, W, bias, nullptr, Out, nullptr);
}

__global__ void __launch_bounds__(256)
out_gemm(const _Float16* __restrict__ A, const _Float16* __restrict__ W,
         const float* __restrict__ bias, const float* __restrict__ resid,
         float* __restrict__ Out) {
  gemm_body<1>(A, W, bias, resid, nullptr, Out);
}

// ---------------------------------------------------------------------------
// Attention: one workgroup per (batch, 16-query tile), loops heads.
// scores -> softmax (f32 stats, f16 P in LDS) -> attn_avg accum -> P·V wmma.
// ---------------------------------------------------------------------------
__global__ void __launch_bounds__(256)
attn_kernel(const _Float16* __restrict__ Qh, const _Float16* __restrict__ Kh,
            const _Float16* __restrict__ Vh, const unsigned char* __restrict__ mask,
            const float* __restrict__ temperature,
            _Float16* __restrict__ Ctx, float* __restrict__ Att) {
  __shared__ __align__(16) _Float16 s_p[16][1032];     // P (and raw scores)
  __shared__ __align__(16) _Float16 s_vt[2][64][40];   // V^T tiles, both halves
  __shared__ __align__(16) float    s_cpart[2][16][64];
  const int tid = threadIdx.x, lane = tid & 31, w = tid >> 5;
  const int q0 = blockIdx.x * 16;
  const int b  = blockIdx.y;
  const int n  = lane & 15, hi = lane >> 4;
  const float temp  = fmaxf(temperature[0], 0.1f);
  const float scale = 1.0f / (8.0f * temp);            // 1/(sqrt(64)*temp)

  for (int h = 0; h < HH; ++h) {
    const _Float16* Qbase = Qh + ((size_t)(b * LL + q0)) * DD + h * HD;
    v16h qa0 = load_a_frag(Qbase,      DD, lane);      // d = 0..31
    v16h qa1 = load_a_frag(Qbase + 32, DD, lane);      // d = 32..63

    // ---- scores: wave w owns key blocks kb = w, w+8, ... (16 keys each)
    for (int kb = w; kb < 64; kb += 8) {
      const _Float16* Kbase = Kh + ((size_t)(b * LL + kb * 16)) * DD + h * HD;
      v16h kf0 = load_bt_frag(Kbase,      DD, lane);
      v16h kf1 = load_bt_frag(Kbase + 32, DD, lane);
      v8f acc = v8f_zero();
      acc = wmma32(qa0, kf0, acc);
      acc = wmma32(qa1, kf1, acc);
      int key = kb * 16 + n;
      bool mk = mask[b * LL + key] != 0;
#pragma unroll
      for (int r = 0; r < 8; ++r) {
        float v = mk ? acc[r] * scale : -65504.0f;
        s_p[r + 8 * hi][key] = (_Float16)v;
      }
    }
    __syncthreads();

    // ---- softmax: wave w handles rows w and w+8
    for (int rr = w; rr < 16; rr += 8) {
      _Float16* row = &s_p[rr][0];
      float mx = -3.0e38f;
      for (int c = lane; c < LL; c += 32) mx = fmaxf(mx, (float)row[c]);
#pragma unroll
      for (int off = 16; off > 0; off >>= 1) mx = fmaxf(mx, __shfl_xor(mx, off, 32));
      float sum = 0.f;
      for (int c = lane; c < LL; c += 32) {
        float e = __expf((float)row[c] - mx);
        row[c] = (_Float16)e;
        sum += e;
      }
#pragma unroll
      for (int off = 16; off > 0; off >>= 1) sum += __shfl_xor(sum, off, 32);
      float inv = 1.0f / sum;
      float* arow = Att + ((size_t)(b * LL + q0 + rr)) * LL;
      for (int c = lane; c < LL; c += 32) {
        float p = (float)row[c] * inv;
        row[c] = (_Float16)p;
        float contrib = p * (1.0f / HH);
        if (h == 0) arow[c] = contrib;
        else        arow[c] += contrib;
      }
    }
    __syncthreads();

    // ---- context = P · V : wave w -> d-tile nt = w&3, key-half = w>>2
    const int nt = w & 3, half = w >> 2;
    v8f cacc = v8f_zero();
    for (int it = 0; it < 16; ++it) {
      {   // stage V^T tiles for both key halves (32 keys x 64 dims each)
        int combo = tid >> 2;            // 0..63 : (half, key)
        int h2s = combo >> 5;            // 0..1
        int key = combo & 31;
        int d0 = (tid & 3) * 16;
        const _Float16* vsrc =
            Vh + ((size_t)(b * LL + h2s * 512 + it * 32 + key)) * DD + h * HD + d0;
#pragma unroll
        for (int j = 0; j < 16; ++j) s_vt[h2s][d0 + j][key] = vsrc[j];
      }
      __syncthreads();
      int key0 = half * 512 + it * 32;
      v16h pa = load_a_frag(&s_p[0][key0], 1032, lane);
      v16h vb = load_bt_frag(&s_vt[half][nt * 16][0], 40, lane);
      cacc = wmma32(pa, vb, cacc);
      __syncthreads();
    }
#pragma unroll
    for (int r = 0; r < 8; ++r) s_cpart[half][r + 8 * hi][nt * 16 + n] = cacc[r];
    __syncthreads();
    for (int i = tid; i < 16 * 64; i += 256) {
      int m = i >> 6, d = i & 63;
      float cv = s_cpart[0][m][d] + s_cpart[1][m][d];
      Ctx[((size_t)(b * LL + q0 + m)) * DD + h * HD + d] = (_Float16)cv;
    }
    __syncthreads();
  }
}

// ---------------------------------------------------------------------------
// LayerNorm over D=1024 per row. One 256-thread block per row.
// ---------------------------------------------------------------------------
__global__ void __launch_bounds__(256)
ln_kernel(const float* __restrict__ R, const float* __restrict__ g,
          const float* __restrict__ beta, float* __restrict__ Out) {
  __shared__ float red[16];
  const int row = blockIdx.x, tid = threadIdx.x, lane = tid & 31, w = tid >> 5;
  const float* x = R + (size_t)row * DD;
  float vals[4];
  float s = 0.f, sq = 0.f;
#pragma unroll
  for (int i = 0; i < 4; ++i) {
    float v = x[tid + i * 256];
    vals[i] = v;
    s += v;
    sq += v * v;
  }
#pragma unroll
  for (int off = 16; off > 0; off >>= 1) {
    s  += __shfl_xor(s, off, 32);
    sq += __shfl_xor(sq, off, 32);
  }
  if (lane == 0) { red[w] = s; red[8 + w] = sq; }
  __syncthreads();
  if (tid < 32) {
    float a  = (tid < 8) ? red[tid] : 0.f;
    float b2 = (tid < 8) ? red[8 + tid] : 0.f;
#pragma unroll
    for (int off = 4; off > 0; off >>= 1) {
      a  += __shfl_xor(a, off, 32);
      b2 += __shfl_xor(b2, off, 32);
    }
    if (tid == 0) { red[0] = a; red[1] = b2; }
  }
  __syncthreads();
  float mean = red[0] * (1.0f / DD);
  float var  = red[1] * (1.0f / DD) - mean * mean;
  float rstd = rsqrtf(var + 1e-5f);
#pragma unroll
  for (int i = 0; i < 4; ++i) {
    int c = tid + i * 256;
    Out[(size_t)row * DD + c] = (vals[i] - mean) * rstd * g[c] + beta[c];
  }
}

// ---------------------------------------------------------------------------
extern "C" void kernel_launch(void* const* d_in, const int* in_sizes, int n_in,
                              void* d_out, int out_size, void* d_ws, size_t ws_size,
                              hipStream_t stream) {
  (void)in_sizes; (void)n_in; (void)out_size; (void)ws_size;
  const float* query = (const float*)d_in[0];
  const float* kv    = (const float*)d_in[1];
  const unsigned char* mask = (const unsigned char*)d_in[2];
  const float* wq = (const float*)d_in[3];
  const float* bq = (const float*)d_in[4];
  const float* wk = (const float*)d_in[5];
  const float* bk = (const float*)d_in[6];
  const float* wv = (const float*)d_in[7];
  const float* bv = (const float*)d_in[8];
  const float* wo = (const float*)d_in[9];
  const float* bo = (const float*)d_in[10];
  const float* lng  = (const float*)d_in[11];
  const float* lnb  = (const float*)d_in[12];
  const float* temp = (const float*)d_in[13];

  float* out = (float*)d_out;                         // (B,L,D)
  float* att = out + (size_t)BB * LL * DD;            // (B,L,L)

  const size_t NE = (size_t)BB * LL * DD;             // 4M
  const size_t NW = (size_t)DD * DD;                  // 1M
  _Float16* Xq  = (_Float16*)d_ws;                    // query  f16
  _Float16* Xkv = Xq  + NE;                           // kv     f16
  _Float16* Wqh = Xkv + NE;
  _Float16* Wkh = Wqh + NW;
  _Float16* Wvh = Wkh + NW;
  _Float16* Woh = Wvh + NW;
  _Float16* Qh  = Woh + NW;
  _Float16* Kh  = Qh  + NE;
  _Float16* Vh  = Kh  + NE;
  _Float16* Ctx = Vh  + NE;
  float*    Rb  = (float*)(Ctx + NE);

  dim3 blk(256);
  // f32 -> f16 pre-conversion (activations + weights)
  cvt_f16<<<dim3(NE / 1024), blk, 0, stream>>>(query, Xq);
  cvt_f16<<<dim3(NE / 1024), blk, 0, stream>>>(kv,    Xkv);
  cvt_f16<<<dim3(NW / 1024), blk, 0, stream>>>(wq, Wqh);
  cvt_f16<<<dim3(NW / 1024), blk, 0, stream>>>(wk, Wkh);
  cvt_f16<<<dim3(NW / 1024), blk, 0, stream>>>(wv, Wvh);
  cvt_f16<<<dim3(NW / 1024), blk, 0, stream>>>(wo, Woh);

  dim3 gp(DD / 64, (BB * LL) / 128);                  // (16, 32)
  proj_gemm<<<gp, blk, 0, stream>>>(Xq,  Wqh, bq, Qh);
  proj_gemm<<<gp, blk, 0, stream>>>(Xkv, Wkh, bk, Kh);
  proj_gemm<<<gp, blk, 0, stream>>>(Xkv, Wvh, bv, Vh);
  attn_kernel<<<dim3(LL / 16, BB), blk, 0, stream>>>(Qh, Kh, Vh, mask, temp, Ctx, att);
  out_gemm<<<gp, blk, 0, stream>>>(Ctx, Woh, bo, query, Rb);
  ln_kernel<<<dim3(BB * LL), blk, 0, stream>>>(Rb, lng, lnb, out);
}
